// Attention_55654186222355
// MI455X (gfx1250) — matrix operand order repaired and checked by reading, not compiled, not run on hardware
//
#include <hip/hip_runtime.h>
#include <hip/hip_bf16.h>

// Problem constants (match reference)
#define B_  2
#define S_  2048
#define D_  1024
#define H_  16
#define HD_ 64

typedef __attribute__((ext_vector_type(16))) __bf16 v16bf;
typedef __attribute__((ext_vector_type(8)))  float  v8f;
typedef __attribute__((ext_vector_type(8)))  __bf16 bf16x8;
typedef __attribute__((ext_vector_type(4)))  unsigned int uint4v;
typedef __attribute__((ext_vector_type(8)))  int int8v;
typedef __attribute__((ext_vector_type(4)))  int int4v;

static __device__ inline v16bf combine8(bf16x8 lo, bf16x8 hi) {
  v16bf r;
#pragma unroll
  for (int e = 0; e < 8; ++e) { r[e] = lo[e]; r[e + 8] = hi[e]; }
  return r;
}

static __device__ inline v8f wmma_bf16(v16bf a, v16bf b, v8f c) {
  // (neg_a, A, neg_b, B, c_mod, C, reuse_a, reuse_b)
  return __builtin_amdgcn_wmma_f32_16x16x32_bf16(false, a, false, b, (short)0, c,
                                                 false, false);
}

// ---------------------------------------------------------------------------
// TDM helper: 2D tile load Global -> LDS via Tensor Data Mover.
// Descriptor layout per CDNA5 ISA (normal mode, 2D):
//   g0: [1:0]=count=1 | [63:32]=lds_addr | [120:64]=global_addr | [127:126]=type=2
//   g1: [17:16]=data_size(=1 -> 2B) | [79:48]=tensor_dim0 | [111:80]=tensor_dim1
//       [127:112]=tile_dim0 | [143:128]=tile_dim1 | [207:160]=tensor_dim0_stride
// Groups 2/3 zero (<=2D tensor).
// ---------------------------------------------------------------------------
static __device__ inline void tdm_load_2d(unsigned long long gaddr,  // bytes
                                          unsigned int lds_addr,     // bytes
                                          unsigned int tensor_d0,    // elems
                                          unsigned int tensor_d1,    // rows
                                          unsigned int tile_d0,      // elems
                                          unsigned int tile_d1,      // rows
                                          unsigned int d0_stride)    // elems
{
#if __has_builtin(__builtin_amdgcn_tensor_load_to_lds)
  uint4v g0;
  g0[0] = 1u;                                        // count=1, user mode
  g0[1] = lds_addr;                                  // lds_addr[31:0]
  g0[2] = (unsigned int)(gaddr & 0xffffffffu);       // global_addr[31:0]
  g0[3] = (unsigned int)((gaddr >> 32) & 0x1ffffffu) // global_addr[56:32]
          | (2u << 30);                              // type = 2 ("image")
  int8v g1;
  g1[0] = (int)(1u << 16);                           // data_size = 1 -> 2 bytes
  g1[1] = (int)((tensor_d0 & 0xffffu) << 16);        // tensor_dim0[15:0] @ bit48
  g1[2] = (int)(((tensor_d0 >> 16) & 0xffffu) |      // tensor_dim0[31:16]
                ((tensor_d1 & 0xffffu) << 16));      // tensor_dim1[15:0] @ bit80
  g1[3] = (int)(((tensor_d1 >> 16) & 0xffffu) |      // tensor_dim1[31:16]
                ((tile_d0 & 0xffffu) << 16));        // tile_dim0 @ bit112
  g1[4] = (int)(tile_d1 & 0xffffu);                  // tile_dim1 @ bit128 (tile_dim2=0)
  g1[5] = (int)d0_stride;                            // tensor_dim0_stride[31:0]
  g1[6] = 0;                                         // stride[47:32] | dim1_stride lo
  g1[7] = 0;
  int4v z4 = {0, 0, 0, 0};
#if defined(__clang_major__) && (__clang_major__ >= 23)
  int8v z8 = {0, 0, 0, 0, 0, 0, 0, 0};
  __builtin_amdgcn_tensor_load_to_lds(g0, g1, z4, z4, z8, 0);
#else
  __builtin_amdgcn_tensor_load_to_lds(g0, g1, z4, z4, 0);
#endif
#else
  (void)gaddr; (void)lds_addr; (void)tensor_d0; (void)tensor_d1;
  (void)tile_d0; (void)tile_d1; (void)d0_stride;
#endif
}

// ---------------------------------------------------------------------------
// Kernel 1: Q/K/V projection (x @ W^T) with fused RoPE on Q,K.
// Each wave owns FOUR 16x16 output tiles sharing one A fragment per K-step
// (4 WMMAs per A load). Q,K stored head-major [bh][s][hd]; V stored
// TRANSPOSED [bh][hd][s] so flash-attention PV B-fragments are contiguous.
// ---------------------------------------------------------------------------
__global__ __launch_bounds__(256) void qkv_rope_kernel(
    const float* __restrict__ x, const int* __restrict__ tpos,
    const float* __restrict__ wq, const float* __restrict__ wk,
    const float* __restrict__ wv,
    __bf16* __restrict__ qws, __bf16* __restrict__ kws, __bf16* __restrict__ vtws)
{
  const int lane = threadIdx.x & 31;
  const int wid  = threadIdx.x >> 5;            // wave in block: 0..7
  const int mt   = blockIdx.x;                  // 0..255  (M tiles over B*S)
  const int ntb  = (blockIdx.y * 8 + wid) * 4;  // base N tile (4 per wave)
  const int z    = blockIdx.z;                  // 0=q 1=k 2=v

  const float* w = (z == 0) ? wq : ((z == 1) ? wk : wv);

  const int col  = lane & 15;
  const int lg   = lane >> 4;
  const int khiA = lg * 8;                      // unused; A frag also uses khiB
  const int khiB = lg * 16;                     // B-frag K offset per lane half
  const int m0   = mt * 16;

  v8f c[4];
#pragma unroll
  for (int j = 0; j < 4; ++j)
#pragma unroll
    for (int r = 0; r < 8; ++r) c[j][r] = 0.f;

  const float* xrow = x + (size_t)(m0 + col) * D_;

  for (int k0 = 0; k0 < D_; k0 += 32) {
    v16bf a;
#pragma unroll
    for (int e = 0; e < 8; ++e) {
      a[e]     = (__bf16)xrow[k0 + khiB + e];          // K = khiB + e
      a[e + 8] = (__bf16)xrow[k0 + khiB + 8 + e];      // K = khiB + 8 + e
    }
#pragma unroll
    for (int j = 0; j < 4; ++j) {
      const float* wrow = w + (size_t)((ntb + j) * 16 + col) * D_;
      v16bf b;
#pragma unroll
      for (int e = 0; e < 16; ++e)
        b[e] = (__bf16)wrow[k0 + khiB + e];            // K = khiB + e
      c[j] = wmma_bf16(a, b, c[j]);
    }
  }

#pragma unroll
  for (int j = 0; j < 4; ++j) {
    const int n  = (ntb + j) * 16 + col;
    const int hd = n & (HD_ - 1);
    const int h  = n >> 6;

    float vals[8];
#pragma unroll
    for (int r = 0; r < 8; ++r) vals[r] = c[j][r];

    if (z < 2) {
      // RoPE: inv_freq = 10000^(-(hd/2)/32)
      const float invf = __expf(-9.2103403719761836f * (float)(hd >> 1) * (1.0f / 32.0f));
      const bool  even = (hd & 1) == 0;
#pragma unroll
      for (int r = 0; r < 8; ++r) {
        const int   m   = m0 + r + lg * 8;
        const int   s   = m & (S_ - 1);
        const float pos = (float)tpos[s];
        const float ang = pos * invf;
        const float sn = __sinf(ang), cs = __cosf(ang);
        const float partner = __shfl_xor(vals[r], 1, 32);  // paired even/odd col
        vals[r] = even ? (vals[r] * cs - partner * sn)
                       : (vals[r] * cs + partner * sn);
      }
    }

#pragma unroll
    for (int r = 0; r < 8; ++r) {
      const int m  = m0 + r + lg * 8;
      const int b_ = m >> 11;            // m / S
      const int s  = m & (S_ - 1);
      const int bh = b_ * H_ + h;
      if (z == 0)
        qws[(((size_t)bh * S_) + s) * HD_ + hd] = (__bf16)vals[r];
      else if (z == 1)
        kws[(((size_t)bh * S_) + s) * HD_ + hd] = (__bf16)vals[r];
      else
        vtws[((size_t)bh * HD_ + hd) * S_ + s] = (__bf16)vals[r];  // V transposed
    }
  }
}

// ---------------------------------------------------------------------------
// Kernel 2: causal flash attention. One block = 8 waves = one (b,h) x 128
// queries. K (32x64) and V^T (64x32) blocks are DMA'd into DOUBLE-BUFFERED
// LDS by the Tensor Data Mover: wave 0 issues block i+1's two TDM loads into
// the alternate buffer, then s_wait_tensorcnt(2) guarantees block i's pair
// is complete (TENSORcnt completes in order per wave) while i+1 stays in
// flight behind the WMMA compute.
// LDS: ksh[2] 8KB @0 | vsh[2] 8KB @8192 | psh 8KB @16384  (dynamic, 24KB)
// ---------------------------------------------------------------------------
__global__ __launch_bounds__(256) void flash_attn_kernel(
    const __bf16* __restrict__ qws, const __bf16* __restrict__ kws,
    const __bf16* __restrict__ vtws, __bf16* __restrict__ aows)
{
  extern __shared__ char smem[];
  __bf16* ksh = (__bf16*)smem;                  // [2][32][64]
  __bf16* vsh = (__bf16*)(smem + 8192);         // [2][64][32]  (hd-major)
  __bf16* psh = (__bf16*)(smem + 16384);        // [8][16][32]

  const int lane = threadIdx.x & 31;
  const int wid  = threadIdx.x >> 5;
  const int bh   = blockIdx.x >> 4;             // (b*H + h)
  const int s0   = (blockIdx.x & 15) * 128;     // query block base
  const int qr0  = s0 + wid * 16;               // this wave's query rows
  const int col  = lane & 15;
  const int lg   = lane >> 4;
  const int khiA = lg * 8;
  const int khiB = lg * 16;
  const size_t bhS = (size_t)bh * S_;
  const int nblk = (s0 + 128) >> 5;             // 32-key blocks to process

  // Q fragments for hd chunks [0,32) and [32,64)
  v16bf aq0, aq1;
  {
    const bf16x8* p = (const bf16x8*)(qws + (bhS + qr0 + col) * HD_);
    aq0 = combine8(p[khiB >> 3], p[(khiB + 8) >> 3]);
    aq1 = combine8(p[(32 + khiB) >> 3], p[(40 + khiB) >> 3]);
  }

  v8f o[4];
#pragma unroll
  for (int t = 0; t < 4; ++t)
#pragma unroll
    for (int r = 0; r < 8; ++r) o[t][r] = 0.f;

  float mrow[8], lrow[8];
#pragma unroll
  for (int r = 0; r < 8; ++r) { mrow[r] = -1e30f; lrow[r] = 0.f; }

#if __has_builtin(__builtin_amdgcn_tensor_load_to_lds)
  // Prologue: stage block 0 into buffer 0.
  if (wid == 0) {
    tdm_load_2d((unsigned long long)(uintptr_t)(kws + bhS * HD_),
                (unsigned int)(uintptr_t)ksh,
                HD_, S_, HD_, 32, HD_);
    tdm_load_2d((unsigned long long)(uintptr_t)(vtws + ((size_t)bh * HD_) * S_),
                (unsigned int)(uintptr_t)vsh,
                S_, HD_, 32, HD_, S_);
  }
#endif

  for (int i = 0; i < nblk; ++i) {
    const int kb = i * 32;
    __syncthreads();                  // alternate buffer's readers are done
#if __has_builtin(__builtin_amdgcn_tensor_load_to_lds)
    if (wid == 0) {
      if (i + 1 < nblk) {
        const int nb = (i + 1) & 1;
        const int knext = kb + 32;
        tdm_load_2d((unsigned long long)(uintptr_t)(kws + (bhS + knext) * HD_),
                    (unsigned int)(uintptr_t)(ksh + nb * (32 * HD_)),
                    HD_, S_, HD_, 32, HD_);
        tdm_load_2d((unsigned long long)(uintptr_t)(vtws + ((size_t)bh * HD_) * S_ + knext),
                    (unsigned int)(uintptr_t)(vsh + nb * (HD_ * 32)),
                    S_, HD_, 32, HD_, S_);
        __builtin_amdgcn_s_wait_tensorcnt(2);   // block i's pair complete
      } else {
        __builtin_amdgcn_s_wait_tensorcnt(0);   // drain on last block
      }
    }
#else
    // Fallback: block-wide cooperative copy of block i (no pipelining)
    {
      const int cb = i & 1;
      for (int e = threadIdx.x; e < 32 * HD_; e += 256) {
        ksh[cb * (32 * HD_) + e] = kws[(bhS + kb) * HD_ + e];
        vsh[cb * (HD_ * 32) + e] =
            vtws[((size_t)bh * HD_ + (e >> 5)) * S_ + kb + (e & 31)];
      }
    }
#endif
    __syncthreads();                  // block i's K/V visible to all waves

    if (kb < qr0 + 16) {              // causal: this wave still has work
      const __bf16* kcur = ksh + (i & 1) * (32 * HD_);
      const __bf16* vcur = vsh + (i & 1) * (HD_ * 32);

      // ---- scores: two 16-key sub-tiles ----
      v8f sc[2];
#pragma unroll
      for (int sub = 0; sub < 2; ++sub) {
        const bf16x8* kp = (const bf16x8*)(kcur + (sub * 16 + col) * HD_);
        v16bf b0 = combine8(kp[khiB >> 3], kp[(khiB + 8) >> 3]);          // hd 0..31
        v16bf b1 = combine8(kp[(32 + khiB) >> 3], kp[(40 + khiB) >> 3]);  // hd 32..63
        v8f t = {0.f, 0.f, 0.f, 0.f, 0.f, 0.f, 0.f, 0.f};
        t = wmma_bf16(aq0, b0, t);
        t = wmma_bf16(aq1, b1, t);
        sc[sub] = t;
      }

      // ---- online softmax (rows live across 16-lane halves) ----
      __bf16* pw = psh + wid * 512;
#pragma unroll
      for (int r = 0; r < 8; ++r) {
        const int mabs = qr0 + r + lg * 8;
        float x0 = sc[0][r] * 0.125f;               // 1/sqrt(64)
        float x1 = sc[1][r] * 0.125f;
        if (kb + col > mabs)      x0 = -1e30f;      // causal mask
        if (kb + 16 + col > mabs) x1 = -1e30f;
        float mx = fmaxf(x0, x1);
#pragma unroll
        for (int d2 = 1; d2 < 16; d2 <<= 1) mx = fmaxf(mx, __shfl_xor(mx, d2, 32));
        const float mnew = fmaxf(mrow[r], mx);
        const float p0 = __expf(x0 - mnew);
        const float p1 = __expf(x1 - mnew);
        float sum = p0 + p1;
#pragma unroll
        for (int d2 = 1; d2 < 16; d2 <<= 1) sum += __shfl_xor(sum, d2, 32);
        const float alpha = __expf(mrow[r] - mnew);
        lrow[r] = lrow[r] * alpha + sum;
        mrow[r] = mnew;
#pragma unroll
        for (int t = 0; t < 4; ++t) o[t][r] *= alpha;
        pw[(r + lg * 8) * 32 + col]      = (__bf16)p0;
        pw[(r + lg * 8) * 32 + 16 + col] = (__bf16)p1;
      }

      __builtin_amdgcn_wave_barrier();  // keep DS store->load ordering tight

      // ---- reload P as A fragment (16x32) ----
      v16bf ap;
      {
        const bf16x8* lp = (const bf16x8*)(pw + col * 32);
        ap = combine8(lp[khiB >> 3], lp[(khiB + 8) >> 3]);
      }

      // ---- PV: 4 hd tiles of 16, B from hd-major vsh (contiguous) ----
#pragma unroll
      for (int t = 0; t < 4; ++t) {
        const bf16x8* vp = (const bf16x8*)(vcur + (t * 16 + col) * 32);
        v16bf bv = combine8(vp[khiB >> 3], vp[(khiB + 8) >> 3]);  // K = khiB + e
        o[t] = wmma_bf16(ap, bv, o[t]);
      }
    }
  }

  // ---- normalize + store (row-major [b][s][D] bf16) ----
  const int b_ = bh >> 4;
  const int h  = bh & 15;
#pragma unroll
  for (int r = 0; r < 8; ++r) {
    const float inv  = 1.0f / lrow[r];
    const int   srow = qr0 + r + lg * 8;
    const size_t base = ((size_t)(b_ * S_ + srow)) * D_ + h * HD_;
#pragma unroll
    for (int t = 0; t < 4; ++t)
      aows[base + t * 16 + col] = (__bf16)(o[t][r] * inv);
  }
}

// ---------------------------------------------------------------------------
// Kernel 3: output projection  out = attn_out @ wo^T  (f32 result).
// 4 N-tiles per wave sharing one A fragment per K-step.
// ---------------------------------------------------------------------------
__global__ __launch_bounds__(256) void out_proj_kernel(
    const __bf16* __restrict__ aows, const float* __restrict__ wo,
    float* __restrict__ out)
{
  const int lane = threadIdx.x & 31;
  const int wid  = threadIdx.x >> 5;
  const int mt   = blockIdx.x;                  // 0..255
  const int ntb  = (blockIdx.y * 8 + wid) * 4;  // base N tile
  const int col  = lane & 15;
  const int lg   = lane >> 4;
  const int khiA = lg * 8;
  const int khiB = lg * 16;
  const int m0 = mt * 16;

  v8f c[4];
#pragma unroll
  for (int j = 0; j < 4; ++j)
#pragma unroll
    for (int r = 0; r < 8; ++r) c[j][r] = 0.f;

  const __bf16* arow = aows + (size_t)(m0 + col) * D_;

  for (int k0 = 0; k0 < D_; k0 += 32) {
    const bf16x8* ap8 = (const bf16x8*)(arow + k0);
    v16bf a = combine8(ap8[khiB >> 3], ap8[(khiB + 8) >> 3]);
#pragma unroll
    for (int j = 0; j < 4; ++j) {
      const float* wrow = wo + (size_t)((ntb + j) * 16 + col) * D_;
      v16bf b;
#pragma unroll
      for (int e = 0; e < 16; ++e)
        b[e] = (__bf16)wrow[k0 + khiB + e];
      c[j] = wmma_bf16(a, b, c[j]);
    }
  }

#pragma unroll
  for (int j = 0; j < 4; ++j)
#pragma unroll
    for (int r = 0; r < 8; ++r)
      out[(size_t)(m0 + r + lg * 8) * D_ + (ntb + j) * 16 + col] = c[j][r];
}

// ---------------------------------------------------------------------------
extern "C" void kernel_launch(void* const* d_in, const int* in_sizes, int n_in,
                              void* d_out, int out_size, void* d_ws, size_t ws_size,
                              hipStream_t stream) {
  (void)in_sizes; (void)n_in; (void)out_size; (void)ws_size;
  const float* x    = (const float*)d_in[0];
  const int*   tpos = (const int*)d_in[1];
  const float* wq   = (const float*)d_in[2];
  const float* wk   = (const float*)d_in[3];
  const float* wv   = (const float*)d_in[4];
  const float* wo   = (const float*)d_in[5];
  float*       out  = (float*)d_out;

  // Workspace (bf16): q | k | v^T | attn_out, each B*H*S*HD = 4M elems (8 MB)
  const size_t per = (size_t)B_ * H_ * S_ * HD_;
  char* wsb = (char*)d_ws;
  __bf16* qws  = (__bf16*)(wsb);
  __bf16* kws  = (__bf16*)(wsb + 2 * per);
  __bf16* vtws = (__bf16*)(wsb + 4 * per);
  __bf16* aows = (__bf16*)(wsb + 6 * per);

  dim3 blk(256);
  // 256 M-tiles x (2 blocks * 8 waves * 4 tiles = 64 N-tiles) x {q,k,v}
  qkv_rope_kernel<<<dim3(256, 2, 3), blk, 0, stream>>>(x, tpos, wq, wk, wv,
                                                       qws, kws, vtws);
  // 32 (b,h) x 16 query-blocks of 128; 24KB dynamic LDS (2x K/V^T + P tiles)
  flash_attn_kernel<<<dim3(512), blk, 24576, stream>>>(qws, kws, vtws, aows);
  // final projection, f32 out
  out_proj_kernel<<<dim3(256, 2), blk, 0, stream>>>(aows, wo, out);
}
